// GNN_graphpred_60730837565599
// MI455X (gfx1250) — compile-verified, matrix operand order checked by reading, not run on hardware
//
#include <hip/hip_runtime.h>
#include <hip/hip_bf16.h>
#include <math.h>

typedef __attribute__((ext_vector_type(16))) _Float16 v16h;
typedef __attribute__((ext_vector_type(8)))  float    v8f;

static constexpr int Nn = 524288;   // nodes
static constexpr int Ee = 1048576;  // edges
static constexpr int Dd = 128;      // emb dim
static constexpr int Gg = 16384;    // graphs

// ---------------------------------------------------------------------------
// helpers
// ---------------------------------------------------------------------------
// mish(x) = x * tanh(softplus(x)).  With t = e^x:
//   tanh(log(1+t)) = ((1+t)^2 - 1)/((1+t)^2 + 1) = t(t+2) / (t(t+2) + 2)
// Clamp x at 20 so t^2 stays finite; ratio saturates to 1 (mish -> x), and
// for x -> -inf the ratio -> t giving the correct x*e^x tail.
// Uses hardware v_exp_f32 + v_rcp_f32 instead of libm tanhf/log1pf.
__device__ __forceinline__ float mish_f(float x) {
    float t = __expf(fminf(x, 20.0f));
    float n = t * (t + 2.0f);
    return x * n * __builtin_amdgcn_rcpf(n + 2.0f);
}

// Load one 16x32 f16 A-fragment for this lane from a 128-float f32 row.
// k0 = kt*32 + kb (kb = 0 for lanes 0-15, 8 for lanes 16-31).
// Lane holds K = k0..k0+7 in halves 0..7 and K = k0+16..k0+23 in halves 8..15.
__device__ __forceinline__ v16h load_a_frag_row(const float* __restrict__ row, int k0) {
    const float* p = row + k0;
    float4 x0 = *(const float4*)(p);
    float4 x1 = *(const float4*)(p + 4);
    float4 x2 = *(const float4*)(p + 16);
    float4 x3 = *(const float4*)(p + 20);
    v16h a;
    a[0]  = (_Float16)x0.x; a[1]  = (_Float16)x0.y; a[2]  = (_Float16)x0.z; a[3]  = (_Float16)x0.w;
    a[4]  = (_Float16)x1.x; a[5]  = (_Float16)x1.y; a[6]  = (_Float16)x1.z; a[7]  = (_Float16)x1.w;
    a[8]  = (_Float16)x2.x; a[9]  = (_Float16)x2.y; a[10] = (_Float16)x2.z; a[11] = (_Float16)x2.w;
    a[12] = (_Float16)x3.x; a[13] = (_Float16)x3.y; a[14] = (_Float16)x3.z; a[15] = (_Float16)x3.w;
    return a;
}

__device__ __forceinline__ v16h load_b_frag(const _Float16* __restrict__ frags,
                                            int kt, int nt, int lane) {
    // fragment (kt,nt): 512 halves; lane's 16 halves are contiguous (32B aligned)
    return *(const v16h*)(frags + (((size_t)(kt * 8 + nt)) << 9) + ((size_t)lane << 4));
}

// ---------------------------------------------------------------------------
// 1) weight prep: f32 [K,128] -> f16 WMMA B fragments
//    frag(kt,nt)[lane*16+h] = (f16) W[(kt*32 + kmap(lane,h)) * 128 + nt*16 + (lane&15)]
// ---------------------------------------------------------------------------
__global__ __launch_bounds__(256) void prep_weights_kernel(
    const float* __restrict__ Wn, const float* __restrict__ We,
    const float* __restrict__ Wp,
    _Float16* __restrict__ wn, _Float16* __restrict__ we, _Float16* __restrict__ wp)
{
    int t = blockIdx.x * blockDim.x + threadIdx.x;
    const int TOTAL = (64 + 64 + 32) * 512;
    if (t >= TOTAL) return;
    int fid = t >> 9;       // global fragment id
    int e   = t & 511;      // element within fragment
    int l   = e >> 4;       // lane 0..31
    int h   = e & 15;       // half 0..15

    const float* W;
    _Float16* dst;
    int f;
    if (fid < 64)       { W = Wn; dst = wn; f = fid; }
    else if (fid < 128) { W = We; dst = we; f = fid - 64; }
    else                { W = Wp; dst = wp; f = fid - 128; }

    int kt = f >> 3;
    int nt = f & 7;
    int kb = (l & 16) ? 8 : 0;
    int k  = kt * 32 + kb + ((h < 8) ? h : (h + 8));
    int n  = nt * 16 + (l & 15);
    dst[((size_t)f << 9) + e] = (_Float16)W[(size_t)k * 128 + n];
}

// ---------------------------------------------------------------------------
// 2) segment sum (atomics) and mean
// ---------------------------------------------------------------------------
__global__ __launch_bounds__(256) void seg_sum_kernel(
    const float* __restrict__ x, const int* __restrict__ batch,
    float* __restrict__ gsum, float* __restrict__ cnt)
{
    int tid  = blockIdx.x * blockDim.x + threadIdx.x;   // N*32 threads
    int node = tid >> 5;
    int c4   = (tid & 31) << 2;
    int b    = batch[node];
    float4 v = *(const float4*)(x + (size_t)node * 128 + c4);
    float* gp = gsum + (size_t)b * 128 + c4;
    unsafeAtomicAdd(gp + 0, v.x);
    unsafeAtomicAdd(gp + 1, v.y);
    unsafeAtomicAdd(gp + 2, v.z);
    unsafeAtomicAdd(gp + 3, v.w);
    if ((tid & 31) == 0) unsafeAtomicAdd(cnt + b, 1.0f);
}

__global__ __launch_bounds__(256) void seg_div_kernel(
    const float* __restrict__ gsum, const float* __restrict__ cnt,
    float* __restrict__ gmean)
{
    int t = blockIdx.x * blockDim.x + threadIdx.x;      // G*D threads
    float c = cnt[t >> 7];
    gmean[t] = gsum[t] / fmaxf(c, 1.0f);
}

// ---------------------------------------------------------------------------
// 3) node MLP + fused self-edge MLP
//    one wave per 16-node M-tile; 8 waves per block
// ---------------------------------------------------------------------------
__global__ __launch_bounds__(256) void node_mlp_kernel(
    const float* __restrict__ x, const int* __restrict__ batch,
    const float* __restrict__ gmean,
    const _Float16* __restrict__ wn, const _Float16* __restrict__ we,
    const float* __restrict__ bn, const float* __restrict__ be,
    float* __restrict__ node_rep, float* __restrict__ self_edge)
{
    __shared__ _Float16 tile[8][16 * 128];   // 32 KB: per-wave f16 node_rep tile

    int lane  = threadIdx.x & 31;
    int wave  = threadIdx.x >> 5;
    int mbase = (blockIdx.x * 8 + wave) * 16;
    int m     = lane & 15;
    int node  = mbase + m;
    int kb    = (lane & 16) ? 8 : 0;
    int rowoff = (lane & 16) ? 8 : 0;
    int col   = lane & 15;

    int gid = batch[node];
    const float* xrow = x + (size_t)node * 128;
    const float* grow = gmean + (size_t)gid * 128;

    // A operand: concat(x[node], g[batch[node]]) as 8 K-tiles of 32
    v16h A[8];
#pragma unroll
    for (int kt = 0; kt < 4; ++kt) A[kt]     = load_a_frag_row(xrow, kt * 32 + kb);
#pragma unroll
    for (int kt = 0; kt < 4; ++kt) A[4 + kt] = load_a_frag_row(grow, kt * 32 + kb);

    // GEMM 1: node_rep = mish(A @ W_node + b_node)
#pragma unroll
    for (int nt = 0; nt < 8; ++nt) {
        v8f c = {};
#pragma unroll
        for (int kt = 0; kt < 8; ++kt) {
            v16h b = load_b_frag(wn, kt, nt, lane);
            c = __builtin_amdgcn_wmma_f32_16x16x32_f16(false, A[kt], false, b,
                                                       (short)0, c, false, false);
        }
        int n = nt * 16 + col;
        float bias = bn[n];
#pragma unroll
        for (int i = 0; i < 8; ++i) {
            float v = mish_f(c[i] + bias);
            int r = rowoff + i;
            node_rep[(size_t)(mbase + r) * 128 + n] = v;
            tile[wave][r * 128 + n] = (_Float16)v;   // stage for self-edge GEMM
        }
    }
    __syncthreads();

    // rebuild A' = concat(node_rep, node_rep) fragments from LDS tile
    v16h A2[4];
#pragma unroll
    for (int j = 0; j < 4; ++j) {
        const _Float16* tp = &tile[wave][m * 128 + j * 32 + kb];
        v16h a;
#pragma unroll
        for (int h = 0; h < 8; ++h) { a[h] = tp[h]; a[8 + h] = tp[16 + h]; }
        A2[j] = a;
    }

    // GEMM 2: self_edge = mish(A' @ W_edge + b_edge); K-tiles repeat mod 4
#pragma unroll
    for (int nt = 0; nt < 8; ++nt) {
        v8f c = {};
#pragma unroll
        for (int kt = 0; kt < 8; ++kt) {
            v16h b = load_b_frag(we, kt, nt, lane);
            c = __builtin_amdgcn_wmma_f32_16x16x32_f16(false, A2[kt & 3], false, b,
                                                       (short)0, c, false, false);
        }
        int n = nt * 16 + col;
        float bias = be[n];
#pragma unroll
        for (int i = 0; i < 8; ++i) {
            float v = mish_f(c[i] + bias);
            self_edge[(size_t)(mbase + rowoff + i) * 128 + n] = v;
        }
    }
}

// ---------------------------------------------------------------------------
// 4) edge MLP with in-register pair averaging
// ---------------------------------------------------------------------------
__global__ __launch_bounds__(256) void edge_mlp_kernel(
    const float* __restrict__ node_rep, const int* __restrict__ ei,
    const _Float16* __restrict__ we, const float* __restrict__ be,
    float* __restrict__ edge_out)
{
    int lane  = threadIdx.x & 31;
    int wave  = threadIdx.x >> 5;
    int mbase = (blockIdx.x * 8 + wave) * 16;   // even -> pairs stay in-tile
    int m     = lane & 15;
    int e     = mbase + m;
    int kb    = (lane & 16) ? 8 : 0;
    int col   = lane & 15;
    int half  = (lane & 16) ? 1 : 0;

    int src = ei[e];
    int dst = ei[Ee + e];
    const float* srow = node_rep + (size_t)src * 128;
    const float* drow = node_rep + (size_t)dst * 128;

    v16h A[8];
#pragma unroll
    for (int kt = 0; kt < 4; ++kt) A[kt]     = load_a_frag_row(srow, kt * 32 + kb);
#pragma unroll
    for (int kt = 0; kt < 4; ++kt) A[4 + kt] = load_a_frag_row(drow, kt * 32 + kb);

#pragma unroll
    for (int nt = 0; nt < 8; ++nt) {
        v8f c = {};
#pragma unroll
        for (int kt = 0; kt < 8; ++kt) {
            v16h b = load_b_frag(we, kt, nt, lane);
            c = __builtin_amdgcn_wmma_f32_16x16x32_f16(false, A[kt], false, b,
                                                       (short)0, c, false, false);
        }
        int n = nt * 16 + col;
        float bias = be[n];
        float mv[8];
#pragma unroll
        for (int i = 0; i < 8; ++i) mv[i] = mish_f(c[i] + bias);
        // C/D layout: lane<16 holds rows 0..7 in c[0..7]; lane>=16 rows 8..15.
        // Edge pair (2k,2k+1) -> adjacent accumulator elements of same lane.
#pragma unroll
        for (int k = 0; k < 4; ++k) {
            float av = 0.5f * (mv[2 * k] + mv[2 * k + 1]);
            int p = (mbase >> 1) + half * 4 + k;
            edge_out[(size_t)p * 128 + n] = av;
        }
    }
}

// ---------------------------------------------------------------------------
// 5) graph_rep = g_mean @ W_pool + b_pool  (K = 128 -> 4 K-tiles)
// ---------------------------------------------------------------------------
__global__ __launch_bounds__(256) void graph_pool_kernel(
    const float* __restrict__ gmean, const _Float16* __restrict__ wp,
    const float* __restrict__ bp, float* __restrict__ graph_rep)
{
    int lane  = threadIdx.x & 31;
    int wave  = threadIdx.x >> 5;
    int mbase = (blockIdx.x * 8 + wave) * 16;
    int m     = lane & 15;
    int kb    = (lane & 16) ? 8 : 0;
    int rowoff = (lane & 16) ? 8 : 0;
    int col   = lane & 15;

    const float* grow = gmean + (size_t)(mbase + m) * 128;
    v16h A[4];
#pragma unroll
    for (int kt = 0; kt < 4; ++kt) A[kt] = load_a_frag_row(grow, kt * 32 + kb);

#pragma unroll
    for (int nt = 0; nt < 8; ++nt) {
        v8f c = {};
#pragma unroll
        for (int kt = 0; kt < 4; ++kt) {
            v16h b = load_b_frag(wp, kt, nt, lane);
            c = __builtin_amdgcn_wmma_f32_16x16x32_f16(false, A[kt], false, b,
                                                       (short)0, c, false, false);
        }
        int n = nt * 16 + col;
        float bias = bp[n];
#pragma unroll
        for (int i = 0; i < 8; ++i)
            graph_rep[(size_t)(mbase + rowoff + i) * 128 + n] = c[i] + bias;
    }
}

// ---------------------------------------------------------------------------
// launch
// ---------------------------------------------------------------------------
extern "C" void kernel_launch(void* const* d_in, const int* in_sizes, int n_in,
                              void* d_out, int out_size, void* d_ws, size_t ws_size,
                              hipStream_t stream) {
    const float* x     = (const float*)d_in[0];
    const int*   batch = (const int*)d_in[1];
    const int*   ei    = (const int*)d_in[2];
    const float* Wn    = (const float*)d_in[3];
    const float* bn    = (const float*)d_in[4];
    const float* We    = (const float*)d_in[5];
    const float* be    = (const float*)d_in[6];
    const float* Wp    = (const float*)d_in[7];
    const float* bp    = (const float*)d_in[8];

    float* out       = (float*)d_out;
    float* node_rep  = out;                                   // [N, D]
    float* self_edge = out + (size_t)Nn * Dd;                 // [N, D]
    float* edge_rep  = out + 2 * (size_t)Nn * Dd;             // [E/2, D]
    float* graph_rep = edge_rep + (size_t)(Ee / 2) * Dd;      // [G, D]

    // workspace layout
    float*    gsum  = (float*)d_ws;                 // [G, D]
    float*    cnt   = gsum + (size_t)Gg * Dd;       // [G]
    float*    gmean = cnt + Gg;                     // [G, D]
    _Float16* wn    = (_Float16*)(gmean + (size_t)Gg * Dd);  // 64 frags * 512
    _Float16* we    = wn + 64 * 512;
    _Float16* wp    = we + 64 * 512;

    hipMemsetAsync(d_ws, 0, (size_t)(Gg * Dd + Gg) * sizeof(float), stream);

    prep_weights_kernel<<<(160 * 512) / 256, 256, 0, stream>>>(Wn, We, Wp, wn, we, wp);

    seg_sum_kernel<<<(Nn * 32) / 256, 256, 0, stream>>>(x, batch, gsum, cnt);
    seg_div_kernel<<<(Gg * Dd) / 256, 256, 0, stream>>>(gsum, cnt, gmean);

    node_mlp_kernel<<<Nn / 16 / 8, 256, 0, stream>>>(x, batch, gmean, wn, we,
                                                     bn, be, node_rep, self_edge);
    edge_mlp_kernel<<<Ee / 16 / 8, 256, 0, stream>>>(node_rep, ei, we, be, edge_rep);
    graph_pool_kernel<<<Gg / 16 / 8, 256, 0, stream>>>(gmean, wp, bp, graph_rep);
}
